// Seq2SeqRNN_85332410236945
// MI455X (gfx1250) — matrix operand order repaired and checked by reading, not compiled, not run on hardware
//
#include <hip/hip_runtime.h>

// Seq2Seq GRU (encoder 128 steps, greedy decoder 64 steps) for MI455X/gfx1250.
//
// MI455X reasoning:
//  * Everything is bandwidth/latency bound (M = B = 32). Weights are converted
//    to bf16 once (~114 MB) so the whole working set is L2-resident (192 MB L2)
//    and per-step traffic is halved; all GEMMs use v_wmma_f32_16x16x32_bf16
//    with f32 accumulation (recurrent state h kept in f32, bf16 shadow for A).
//  * The encoder's x@Wih.T gates are input-independent across time -> hoisted
//    into ONE parallel WMMA GEMM over M = S*B = 4096 rows. The 128 dependent
//    step kernels then only run the h@Whh.T recurrence (shorter critical path).
//  * Decoder logits GEMM dominates traffic (64 x 65.5 MB bf16 from L2). Each
//    wave owns TWO N-tiles sharing one A tile: 3 b128 loads per WMMA instead
//    of 4, plus a global_prefetch one K-block ahead on the weight stream.

#define DEV_INLINE __device__ __forceinline__

typedef __attribute__((ext_vector_type(16))) __bf16 v16bf;
typedef __attribute__((ext_vector_type(8)))  __bf16 v8bf;
typedef __attribute__((ext_vector_type(8)))  float  v8f;

static constexpr int B_ = 32;
static constexpr int S_ = 128;
static constexpr int T_ = 64;
static constexpr int H_ = 1024;
static constexpr int E_ = 256;
static constexpr int V_ = 32000;
static constexpr int SOS_ = 1;

DEV_INLINE v8f wmma_bf16(v16bf a, v16bf b, v8f c) {
  // (neg_a, A, neg_b, B, c_mod, C, reuse_a, reuse_b)
  return __builtin_amdgcn_wmma_f32_16x16x32_bf16(false, a, false, b, (short)0, c,
                                                 false, false);
}

// A-matrix (16x32 bf16): lane supplies row M = lane&15.
// lane<16 : elements 0..15 = K[0..16)
// lane>=16: elements 0..15 = K[16..32)
// caller passes p already offset by (kk + (lane>=16 ? 16 : 0)); chunks at +0,+8.
DEV_INLINE v16bf load_a_tile(const __bf16* p) {
  union { v16bf v; v8bf h[2]; } u;
  u.h[0] = *(const v8bf*)(p);
  u.h[1] = *(const v8bf*)(p + 8);
  return u.v;
}

// B-matrix (32x16 bf16): lane = column N (lane&15); lane<16 holds K[0..16),
// lane>=16 holds K[16..32) -> 16 contiguous K values = one 32B load.
DEV_INLINE v16bf load_b_tile(const __bf16* p) {
  return *(const v16bf*)(p);
}

DEV_INLINE float sigmoidf_(float x) { return 1.0f / (1.0f + __expf(-x)); }

DEV_INLINE v8f zero_v8f() {
  v8f a;
#pragma unroll
  for (int i = 0; i < 8; ++i) a[i] = 0.0f;
  return a;
}

// ---------------------------------------------------------------------------
// Encoder input gates, all timesteps at once (fully parallel WMMA GEMM):
//   gi[t*B + b, :] = embed[src[b,t]] @ enc_Wih.T        shape [S*B, 3H]
// Grid: (3H/16 = 192) x (S*B/32 = 128), 64 threads = 2 waves (M-tiles).
// ---------------------------------------------------------------------------
__global__ void __launch_bounds__(64)
enc_gates_kernel(const int* __restrict__ src,       // [B,S]
                 const __bf16* __restrict__ embed,  // [V,E]
                 const __bf16* __restrict__ Wih,    // [3H,E]
                 float* __restrict__ gi) {          // [S*B, 3H]
  const int wave = threadIdx.x >> 5;
  const int lane = threadIdx.x & 31;
  const int c = lane & 15;
  const bool hi = lane >= 16;
  const int abase = hi ? 16 : 0;
  const int bbase = hi ? 16 : 0;
  const int g0 = blockIdx.y * 32 + wave * 16;  // row-tile base in [0, S*B)
  const int col = blockIdx.x * 16 + c;

  // A row for this lane: g = t*B + b  (B = 32 -> t = g>>5, b = g&31)
  const int g = g0 + c;
  const int t = g >> 5;
  const int b = g & 31;
  const int tok = src[b * S_ + t];
  const __bf16* xrow = embed + (long long)tok * E_;

  const __bf16* wr0 = Wih + (long long)(0 * H_ + col) * E_;
  const __bf16* wz0 = Wih + (long long)(1 * H_ + col) * E_;
  const __bf16* wn0 = Wih + (long long)(2 * H_ + col) * E_;

  v8f accR = zero_v8f(), accZ = zero_v8f(), accN = zero_v8f();
#pragma unroll
  for (int kk = 0; kk < E_; kk += 32) {
    v16bf a = load_a_tile(xrow + kk + abase);
    accR = wmma_bf16(a, load_b_tile(wr0 + kk + bbase), accR);
    accZ = wmma_bf16(a, load_b_tile(wz0 + kk + bbase), accZ);
    accN = wmma_bf16(a, load_b_tile(wn0 + kk + bbase), accN);
  }
#pragma unroll
  for (int i = 0; i < 8; ++i) {
    const int grow = g0 + i + (hi ? 8 : 0);
    float* o = gi + (long long)grow * (3 * H_);
    o[col] = accR[i];
    o[H_ + col] = accZ[i];
    o[2 * H_ + col] = accN[i];
  }
}

// ---------------------------------------------------------------------------
// One fused GRU step. Grid: H/16 = 64 workgroups (16 output cols each),
// 64 threads = 2 waves (wave w -> batch rows 16w..16w+15).
// If gi_t != null (encoder): input gates come precomputed -> only the
// h@Whh.T recurrence runs here. Else (decoder): gather embedding and fuse
// the K=E input-gate GEMM too.
// ---------------------------------------------------------------------------
__global__ void __launch_bounds__(64)
gru_step_kernel(const float* __restrict__ gi_t,     // [B,3H] or null
                const int* __restrict__ idx,        // [B] tokens (decoder)
                const __bf16* __restrict__ embed,   // [V,E]
                const __bf16* __restrict__ Wih,     // [3H,E]
                const __bf16* __restrict__ Whh,     // [3H,H]
                const float* __restrict__ bih, const float* __restrict__ bhh,
                const float* __restrict__ h_in,     // [B,H] f32
                const __bf16* __restrict__ h_in_bf, // [B,H] bf16
                float* __restrict__ h_out, __bf16* __restrict__ h_out_bf,
                const int* __restrict__ src_len,    // null => always update
                int t) {
  const int wave = threadIdx.x >> 5;
  const int lane = threadIdx.x & 31;
  const int m0 = wave * 16;               // batch-row tile base
  const int c = lane & 15;
  const int col = blockIdx.x * 16 + c;    // hidden column this lane owns
  const bool hi = lane >= 16;
  const int abase = hi ? 16 : 0;
  const int bbase = hi ? 16 : 0;

  v8f accR = zero_v8f(), accZ = zero_v8f(), accIN = zero_v8f(), accHN = zero_v8f();

  const int arow = m0 + c;                // batch row this lane feeds into A

  if (gi_t != nullptr) {
    // Precomputed input gates (encoder): 24 scalar loads instead of a K=E GEMM.
#pragma unroll
    for (int i = 0; i < 8; ++i) {
      const int row = m0 + i + (hi ? 8 : 0);
      const float* go = gi_t + (long long)row * (3 * H_);
      accR[i] = go[col];
      accZ[i] = go[H_ + col];
      accIN[i] = go[2 * H_ + col];
    }
  } else {
    // Decoder: gather embedding of data-dependent token, fuse x@Wih.T.
    const int tok = idx[arow];
    const __bf16* xrow = embed + (long long)tok * E_;
    const __bf16* wr0 = Wih + (long long)(0 * H_ + col) * E_;
    const __bf16* wz0 = Wih + (long long)(1 * H_ + col) * E_;
    const __bf16* wn0 = Wih + (long long)(2 * H_ + col) * E_;
#pragma unroll
    for (int kk = 0; kk < E_; kk += 32) {
      v16bf a = load_a_tile(xrow + kk + abase);
      accR  = wmma_bf16(a, load_b_tile(wr0 + kk + bbase), accR);
      accZ  = wmma_bf16(a, load_b_tile(wz0 + kk + bbase), accZ);
      accIN = wmma_bf16(a, load_b_tile(wn0 + kk + bbase), accIN);
    }
  }

  // ---- h @ Whh.T contribution (the serial part) ----
  const __bf16* hrow = h_in_bf + (long long)arow * H_;
  const __bf16* ur0 = Whh + (long long)(0 * H_ + col) * H_;
  const __bf16* uz0 = Whh + (long long)(1 * H_ + col) * H_;
  const __bf16* un0 = Whh + (long long)(2 * H_ + col) * H_;
  for (int kk = 0; kk < H_; kk += 32) {
    v16bf a = load_a_tile(hrow + kk + abase);
    accR  = wmma_bf16(a, load_b_tile(ur0 + kk + bbase), accR);
    accZ  = wmma_bf16(a, load_b_tile(uz0 + kk + bbase), accZ);
    accHN = wmma_bf16(a, load_b_tile(un0 + kk + bbase), accHN);
  }

  // ---- gates + state update (this lane owns a fixed column) ----
  const float br_i = bih[col],          br_h = bhh[col];
  const float bz_i = bih[H_ + col],     bz_h = bhh[H_ + col];
  const float bn_i = bih[2 * H_ + col], bn_h = bhh[2 * H_ + col];
#pragma unroll
  for (int i = 0; i < 8; ++i) {
    const int row = m0 + i + (hi ? 8 : 0);   // C/D layout: lanes>=16 -> M=8+i
    float r = sigmoidf_(accR[i] + br_i + br_h);
    float z = sigmoidf_(accZ[i] + bz_i + bz_h);
    float n = tanhf(accIN[i] + bn_i + r * (accHN[i] + bn_h));
    const long long o = (long long)row * H_ + col;
    float hp = h_in[o];
    float hn = (1.0f - z) * n + z * hp;
    if (src_len != nullptr && t >= src_len[row]) hn = hp;  // packed-seq freeze
    h_out[o] = hn;
    h_out_bf[o] = (__bf16)hn;
  }
}

// ---------------------------------------------------------------------------
// logits[b, t, :] = h @ dec_W.T + dec_b.
// Grid: V/32 = 1000 WGs, 64 threads = 2 waves (M-tiles). Each wave owns TWO
// N-tiles sharing one A tile: 3 b128 loads per WMMA. Weight stream prefetched
// one K-block ahead (global_prefetch).
// ---------------------------------------------------------------------------
__global__ void __launch_bounds__(64)
logits_kernel(const __bf16* __restrict__ h_bf,   // [B,H] bf16
              const __bf16* __restrict__ Wout,   // [V,H] bf16
              const float* __restrict__ bout,    // [V]
              float* __restrict__ out,           // [B,T,V]
              int t) {
  const int wave = threadIdx.x >> 5;
  const int lane = threadIdx.x & 31;
  const int m0 = wave * 16;
  const int c = lane & 15;
  const int col0 = blockIdx.x * 32 + c;
  const int col1 = col0 + 16;
  const bool hi = lane >= 16;
  const int abase = hi ? 16 : 0;
  const int bbase = hi ? 16 : 0;

  v8f acc0 = zero_v8f(), acc1 = zero_v8f();
  const __bf16* hrow = h_bf + (long long)(m0 + c) * H_;
  const __bf16* w0 = Wout + (long long)col0 * H_;
  const __bf16* w1 = Wout + (long long)col1 * H_;
  for (int kk = 0; kk < H_; kk += 32) {
    if (kk + 32 < H_) {  // stream hint one K-block ahead on the weight rows
      __builtin_prefetch(w0 + kk + 32 + bbase, 0, 1);
      __builtin_prefetch(w1 + kk + 32 + bbase, 0, 1);
    }
    v16bf a = load_a_tile(hrow + kk + abase);
    acc0 = wmma_bf16(a, load_b_tile(w0 + kk + bbase), acc0);
    acc1 = wmma_bf16(a, load_b_tile(w1 + kk + bbase), acc1);
  }
  const float bb0 = bout[col0];
  const float bb1 = bout[col1];
#pragma unroll
  for (int i = 0; i < 8; ++i) {
    const int row = m0 + i + (hi ? 8 : 0);
    float* orow = out + ((long long)row * T_ + t) * V_;
    orow[col0] = acc0[i] + bb0;
    orow[col1] = acc1[i] + bb1;
  }
}

// ---------------------------------------------------------------------------
// Greedy argmax over V per batch row (first-occurrence tie-break like jnp).
// Grid: B blocks, 256 threads.
// ---------------------------------------------------------------------------
__global__ void __launch_bounds__(256)
argmax_kernel(const float* __restrict__ logits, int t, int* __restrict__ inp) {
  __shared__ float sv[256];
  __shared__ int si[256];
  const int b = blockIdx.x;
  const float* row = logits + ((long long)b * T_ + t) * V_;
  float best = -3.402823466e38f;
  int bi = 0;
  for (int v = threadIdx.x; v < V_; v += 256) {
    float x = row[v];
    if (x > best) { best = x; bi = v; }   // v ascending -> first max kept
  }
  sv[threadIdx.x] = best;
  si[threadIdx.x] = bi;
  __syncthreads();
  for (int s = 128; s > 0; s >>= 1) {
    if (threadIdx.x < s) {
      float ov = sv[threadIdx.x + s];
      int oi = si[threadIdx.x + s];
      if (ov > sv[threadIdx.x] ||
          (ov == sv[threadIdx.x] && oi < si[threadIdx.x])) {
        sv[threadIdx.x] = ov;
        si[threadIdx.x] = oi;
      }
    }
    __syncthreads();
  }
  if (threadIdx.x == 0) inp[b] = si[0];
}

// ---------------------------------------------------------------------------
// One-time (per launch) f32 -> bf16 conversion; makes the whole weight working
// set (~114 MB) L2-resident on MI455X's 192MB L2 and halves per-step traffic.
// ---------------------------------------------------------------------------
__global__ void __launch_bounds__(256)
cvt_bf16_kernel(const float* __restrict__ src, __bf16* __restrict__ dst,
                long long n) {
  long long i = (long long)blockIdx.x * 256 + threadIdx.x;
  const long long stride = (long long)gridDim.x * 256;
  for (; i < n; i += stride) dst[i] = (__bf16)src[i];
}

__global__ void __launch_bounds__(256)
init_state_kernel(float* __restrict__ h_f, __bf16* __restrict__ h_b,
                  int* __restrict__ inp) {
  const int i = blockIdx.x * 256 + threadIdx.x;
  if (i < B_ * H_) {
    h_f[i] = 0.0f;
    h_b[i] = (__bf16)0.0f;
  }
  if (i < B_) inp[i] = SOS_;
}

extern "C" void kernel_launch(void* const* d_in, const int* in_sizes, int n_in,
                              void* d_out, int out_size, void* d_ws,
                              size_t ws_size, hipStream_t stream) {
  (void)in_sizes; (void)n_in; (void)out_size; (void)ws_size;

  const int*   src       = (const int*)d_in[0];
  /* d_in[1] = tgt, unused in eval (teacher_forcing_p = 0) */
  const int*   src_len   = (const int*)d_in[2];
  const float* enc_embed = (const float*)d_in[3];
  const float* enc_Wih   = (const float*)d_in[4];
  const float* enc_Whh   = (const float*)d_in[5];
  const float* enc_bih   = (const float*)d_in[6];
  const float* enc_bhh   = (const float*)d_in[7];
  const float* dec_embed = (const float*)d_in[8];
  const float* dec_Wih   = (const float*)d_in[9];
  const float* dec_Whh   = (const float*)d_in[10];
  const float* dec_bih   = (const float*)d_in[11];
  const float* dec_bhh   = (const float*)d_in[12];
  const float* dec_W     = (const float*)d_in[13];
  const float* dec_b     = (const float*)d_in[14];
  float* out = (float*)d_out;

  // ---- workspace carve-up (~165 MB, 256B-aligned blocks) ----
  char* p = (char*)d_ws;
  auto take = [&](size_t bytes) -> void* {
    void* r = (void*)p;
    p += (bytes + 255) & ~(size_t)255;
    return r;
  };
  __bf16* enc_embed_bf = (__bf16*)take((size_t)V_ * E_ * 2);
  __bf16* enc_Wih_bf   = (__bf16*)take((size_t)3 * H_ * E_ * 2);
  __bf16* enc_Whh_bf   = (__bf16*)take((size_t)3 * H_ * H_ * 2);
  __bf16* dec_embed_bf = (__bf16*)take((size_t)V_ * E_ * 2);
  __bf16* dec_Wih_bf   = (__bf16*)take((size_t)3 * H_ * E_ * 2);
  __bf16* dec_Whh_bf   = (__bf16*)take((size_t)3 * H_ * H_ * 2);
  __bf16* dec_W_bf     = (__bf16*)take((size_t)V_ * H_ * 2);
  float*  gi_enc       = (float*)take((size_t)S_ * B_ * 3 * H_ * 4);  // 50.3MB
  float*  hf[2];
  __bf16* hb[2];
  hf[0] = (float*)take((size_t)B_ * H_ * 4);
  hf[1] = (float*)take((size_t)B_ * H_ * 4);
  hb[0] = (__bf16*)take((size_t)B_ * H_ * 2);
  hb[1] = (__bf16*)take((size_t)B_ * H_ * 2);
  int* inp = (int*)take((size_t)B_ * 4);

  // ---- one-time precision conversion (deterministic, re-run every call) ----
  cvt_bf16_kernel<<<2048, 256, 0, stream>>>(enc_embed, enc_embed_bf,
                                            (long long)V_ * E_);
  cvt_bf16_kernel<<<512, 256, 0, stream>>>(enc_Wih, enc_Wih_bf,
                                           (long long)3 * H_ * E_);
  cvt_bf16_kernel<<<2048, 256, 0, stream>>>(enc_Whh, enc_Whh_bf,
                                            (long long)3 * H_ * H_);
  cvt_bf16_kernel<<<2048, 256, 0, stream>>>(dec_embed, dec_embed_bf,
                                            (long long)V_ * E_);
  cvt_bf16_kernel<<<512, 256, 0, stream>>>(dec_Wih, dec_Wih_bf,
                                           (long long)3 * H_ * E_);
  cvt_bf16_kernel<<<2048, 256, 0, stream>>>(dec_Whh, dec_Whh_bf,
                                            (long long)3 * H_ * H_);
  cvt_bf16_kernel<<<4096, 256, 0, stream>>>(dec_W, dec_W_bf,
                                            (long long)V_ * H_);

  init_state_kernel<<<(B_ * H_ + 255) / 256, 256, 0, stream>>>(hf[0], hb[0],
                                                               inp);

  // ---- encoder input gates for ALL timesteps: one parallel WMMA GEMM ----
  {
    dim3 grid(3 * H_ / 16, (S_ * B_) / 32);  // 192 x 128
    enc_gates_kernel<<<grid, 64, 0, stream>>>(src, enc_embed_bf, enc_Wih_bf,
                                              gi_enc);
  }

  int cur = 0;
  // ---- encoder: 128 dependent recurrence-only steps, h ping-pong ----
  for (int t = 0; t < S_; ++t) {
    gru_step_kernel<<<H_ / 16, 64, 0, stream>>>(
        gi_enc + (long long)t * B_ * 3 * H_, nullptr, enc_embed_bf, enc_Wih_bf,
        enc_Whh_bf, enc_bih, enc_bhh, hf[cur], hb[cur], hf[1 - cur],
        hb[1 - cur], src_len, t);
    cur ^= 1;
  }

  // ---- decoder: greedy autoregressive loop ----
  for (int t = 0; t < T_; ++t) {
    gru_step_kernel<<<H_ / 16, 64, 0, stream>>>(
        nullptr, inp, dec_embed_bf, dec_Wih_bf, dec_Whh_bf, dec_bih, dec_bhh,
        hf[cur], hb[cur], hf[1 - cur], hb[1 - cur], nullptr, t);
    cur ^= 1;
    logits_kernel<<<V_ / 32, 64, 0, stream>>>(hb[cur], dec_W_bf, dec_b, out, t);
    argmax_kernel<<<B_, 256, 0, stream>>>(out, t, inp);
  }
}
